// Atten_65163243815331
// MI455X (gfx1250) — compile-verified, matrix operand order checked by reading, not run on hardware
//
#include <hip/hip_runtime.h>

typedef _Float16 half_t;
typedef __attribute__((ext_vector_type(16))) _Float16 v16h;
typedef __attribute__((ext_vector_type(8)))  _Float16 v8h;
typedef __attribute__((ext_vector_type(8)))  float    v8f;
typedef __attribute__((ext_vector_type(4)))  float    v4f;

#define BB   32
#define CC   512
#define SS   1024
#define NG   32
#define CPG  16
#define ATT_SCALE 0.044194173824159216f

// ---------------------------------------------------------------------------
// WMMA helpers (wave32, 16x16x32 f16 -> f32)
// A fragment: lane L (r=L&15, hi=L>>4) holds row (row_base+r),
//             halves  K = kb + hi*8 + {0..7}      (VGPR 0..3)
//             and     K = kb + 16 + hi*8 + {0..7} (VGPR 4..7)
// B fragment uses the same per-lane layout with "row" = output column,
// reading rows of W (columns of W^T), so both loads are row-major.
// ---------------------------------------------------------------------------
__device__ __forceinline__ v16h load_frag(const half_t* __restrict__ base,
                                          long ld, int row_base, int kb, int lane) {
  const int r  = lane & 15;
  const int hi = lane >> 4;
  const half_t* p = base + (long)(row_base + r) * ld + (kb + hi * 8);
  v8h lo = *(const v8h*)p;
  v8h hh = *(const v8h*)(p + 16);
  v16h out;
#pragma unroll
  for (int i = 0; i < 8; ++i) { out[i] = lo[i]; out[8 + i] = hh[i]; }
  return out;
}

__device__ __forceinline__ v8f wmma16(v16h a, v16h b, v8f c) {
  return __builtin_amdgcn_wmma_f32_16x16x32_f16(
      /*neg_a=*/false, a, /*neg_b=*/false, b,
      /*c_mod=*/(short)0, c, /*reuse_a=*/false, /*reuse_b=*/false);
}

// 4x2 register-blocked GEMM core: one wave computes a 64x32 output tile.
// Per K-step: 4 A frags + 2 B frags (12x b128) feed 8 WMMAs.
__device__ __forceinline__ void gemm4x2(const half_t* __restrict__ A, long lda, int m0,
                                        const half_t* __restrict__ Bm, long ldb, int n0,
                                        int K, int lane, v8f acc[8]) {
  for (int kb = 0; kb < K; kb += 32) {
    v16h b0 = load_frag(Bm, ldb, n0,      kb, lane);
    v16h b1 = load_frag(Bm, ldb, n0 + 16, kb, lane);
#pragma unroll
    for (int i = 0; i < 4; ++i) {
      v16h a = load_frag(A, lda, m0 + i * 16, kb, lane);
      acc[i * 2 + 0] = wmma16(a, b0, acc[i * 2 + 0]);
      acc[i * 2 + 1] = wmma16(a, b1, acc[i * 2 + 1]);
    }
  }
}

// ---------------------------------------------------------------------------
// 1) fp32 -> f16 weight conversion (Wq, Wk, Wv, Wo)
// ---------------------------------------------------------------------------
__global__ __launch_bounds__(256) void wcvt_kernel(const float* __restrict__ Wq,
                                                   const float* __restrict__ Wk,
                                                   const float* __restrict__ Wv,
                                                   const float* __restrict__ Wo,
                                                   half_t* __restrict__ dst) {
  const int m = blockIdx.y;
  const float* s = (m == 0) ? Wq : (m == 1) ? Wk : (m == 2) ? Wv : Wo;
  long i = (long)blockIdx.x * blockDim.x + threadIdx.x;   // < 512*512
  dst[(long)m * CC * CC + i] = (half_t)s[i];
}

// ---------------------------------------------------------------------------
// 2) GroupNorm: one block per (batch, group); writes t[b*S + s, c] in f16
// ---------------------------------------------------------------------------
__global__ __launch_bounds__(256) void gnorm_kernel(const float* __restrict__ x,
                                                    const float* __restrict__ gw,
                                                    const float* __restrict__ gb,
                                                    half_t* __restrict__ t16) {
  __shared__ float s_sum[256];
  __shared__ float s_sq[256];
  const int b = blockIdx.x >> 5;
  const int g = blockIdx.x & 31;
  const int tid = threadIdx.x;
  const long base = ((long)b * CC + g * CPG) * SS;   // x[b, g*16, 0]
  float sum = 0.f, sq = 0.f;
  for (int idx = tid; idx < CPG * SS; idx += 256) {
    float v = x[base + idx];
    sum += v; sq += v * v;
  }
  s_sum[tid] = sum; s_sq[tid] = sq;
  __syncthreads();
  for (int off = 128; off > 0; off >>= 1) {
    if (tid < off) { s_sum[tid] += s_sum[tid + off]; s_sq[tid] += s_sq[tid + off]; }
    __syncthreads();
  }
  const float inv_n = 1.0f / (float)(CPG * SS);
  const float mean  = s_sum[0] * inv_n;
  const float var   = s_sq[0] * inv_n - mean * mean;
  const float rs    = rsqrtf(var + 1e-6f);
  for (int idx = tid; idx < CPG * SS; idx += 256) {
    const int cl = idx >> 10;          // channel within group
    const int s  = idx & 1023;         // spatial
    const int c  = g * CPG + cl;
    float v = (x[base + idx] - mean) * rs * gw[c] + gb[c];
    t16[((long)b * SS + s) * CC + c] = (half_t)v;
  }
}

// ---------------------------------------------------------------------------
// 3) Fused QKV GEMM: [32768,512] x [512,512]^T (+bias) for q,k,v.
//    q,k stored row-major f16; v stored transposed [B, C, S] f16.
//    One wave per 64x32 output tile (4x2 WMMA blocking).
// ---------------------------------------------------------------------------
__global__ __launch_bounds__(256) void qkv_kernel(const half_t* __restrict__ t16,
                                                  const half_t* __restrict__ w16,
                                                  const float* __restrict__ bq,
                                                  const float* __restrict__ bk,
                                                  const float* __restrict__ bv,
                                                  half_t* __restrict__ q16,
                                                  half_t* __restrict__ k16,
                                                  half_t* __restrict__ vT16) {
  const int lane = threadIdx.x & 31;
  const int wave = threadIdx.x >> 5;
  const long tile = (long)blockIdx.x * 8 + wave;       // < 3*8192
  const int  mat  = (int)(tile >> 13);                 // 0=q 1=k 2=v
  const int  rem  = (int)(tile & 8191);
  const int  nt   = rem & 15;                          // 16 n-tiles of 32
  const int  mt   = rem >> 4;                          // 512 m-tiles of 64
  const int  m0   = mt * 64;
  const int  n0   = nt * 32;
  const half_t* wsel = w16 + (long)mat * CC * CC;
  const float*  bsel = (mat == 0) ? bq : (mat == 1) ? bk : bv;

  v8f acc[8] = {};
  gemm4x2(t16, CC, m0, wsel, CC, n0, CC, lane, acc);

  const int hi = lane >> 4;
  const int nl = lane & 15;
  if (mat < 2) {
    half_t* dst = (mat == 0) ? q16 : k16;
#pragma unroll
    for (int i = 0; i < 4; ++i) {
#pragma unroll
      for (int j = 0; j < 2; ++j) {
        const int   n    = n0 + j * 16 + nl;
        const float bias = bsel[n];
        const v8f   a    = acc[i * 2 + j];
#pragma unroll
        for (int r = 0; r < 8; ++r) {
          long m = (long)m0 + i * 16 + hi * 8 + r;
          dst[m * CC + n] = (half_t)(a[r] + bias);
        }
      }
    }
  } else {
#pragma unroll
    for (int i = 0; i < 4; ++i) {
      const long m0i = (long)m0 + i * 16 + hi * 8;     // 8 consecutive tokens
      const int  b   = (int)(m0i >> 10);
      const int  s0  = (int)(m0i & 1023);
#pragma unroll
      for (int j = 0; j < 2; ++j) {
        const int   n    = n0 + j * 16 + nl;
        const float bias = bsel[n];
        const v8f   a    = acc[i * 2 + j];
        v8h pack;
#pragma unroll
        for (int r = 0; r < 8; ++r) pack[r] = (half_t)(a[r] + bias);
        *(v8h*)(vT16 + ((long)b * CC + n) * SS + s0) = pack;
      }
    }
  }
}

// ---------------------------------------------------------------------------
// 4) Scores: per batch S = q k^T * SCALE  -> f16 [b, s, t]
// ---------------------------------------------------------------------------
__global__ __launch_bounds__(256) void scores_kernel(const half_t* __restrict__ q16,
                                                     const half_t* __restrict__ k16,
                                                     half_t* __restrict__ p16) {
  const int lane = threadIdx.x & 31;
  const int wave = threadIdx.x >> 5;
  const long tile = (long)blockIdx.x * 8 + wave;       // < 32*512
  const int  b   = (int)(tile >> 9);
  const int  rem = (int)(tile & 511);
  const int  mt  = rem >> 5;                           // 16 query tiles of 64
  const int  nt  = rem & 31;                           // 32 key tiles of 32
  const int  m0  = mt * 64;
  const int  n0  = nt * 32;

  v8f acc[8] = {};
  gemm4x2(q16, CC, b * SS + m0, k16, CC, b * SS + n0, CC, lane, acc);

  const int hi = lane >> 4;
  const int nl = lane & 15;
  const long base = (long)b * SS * SS;
#pragma unroll
  for (int i = 0; i < 4; ++i) {
#pragma unroll
    for (int j = 0; j < 2; ++j) {
      const int n = n0 + j * 16 + nl;
      const v8f a = acc[i * 2 + j];
#pragma unroll
      for (int r = 0; r < 8; ++r) {
        const int m = m0 + i * 16 + hi * 8 + r;
        p16[base + (long)m * SS + n] = (half_t)(a[r] * ATT_SCALE);
      }
    }
  }
}

// ---------------------------------------------------------------------------
// 5) Row softmax over 1024 keys, in place on f16 scores (math in f32)
// ---------------------------------------------------------------------------
__global__ __launch_bounds__(256) void softmax_kernel(half_t* __restrict__ p16) {
  __shared__ float red[256];
  half_t* p = p16 + (long)blockIdx.x * SS;
  const int tid = threadIdx.x;
  float v[4];
#pragma unroll
  for (int i = 0; i < 4; ++i) v[i] = (float)p[tid + i * 256];
  float mx = fmaxf(fmaxf(v[0], v[1]), fmaxf(v[2], v[3]));
  red[tid] = mx;
  __syncthreads();
  for (int off = 128; off > 0; off >>= 1) {
    if (tid < off) red[tid] = fmaxf(red[tid], red[tid + off]);
    __syncthreads();
  }
  mx = red[0];
  __syncthreads();
  float e[4], s = 0.f;
#pragma unroll
  for (int i = 0; i < 4; ++i) { e[i] = __expf(v[i] - mx); s += e[i]; }
  red[tid] = s;
  __syncthreads();
  for (int off = 128; off > 0; off >>= 1) {
    if (tid < off) red[tid] += red[tid + off];
    __syncthreads();
  }
  const float inv = 1.0f / red[0];
#pragma unroll
  for (int i = 0; i < 4; ++i) p[tid + i * 256] = (half_t)(e[i] * inv);
}

// ---------------------------------------------------------------------------
// 6) O = P @ V  (per batch: [1024,1024] x [1024,512]); V read from vT rows
// ---------------------------------------------------------------------------
__global__ __launch_bounds__(256) void pv_kernel(const half_t* __restrict__ p16,
                                                 const half_t* __restrict__ vT16,
                                                 half_t* __restrict__ o16) {
  const int lane = threadIdx.x & 31;
  const int wave = threadIdx.x >> 5;
  const long tile = (long)blockIdx.x * 8 + wave;       // < 32*256
  const int  b   = (int)(tile >> 8);
  const int  rem = (int)(tile & 255);
  const int  mt  = rem >> 4;                           // 16 token tiles of 64
  const int  nt  = rem & 15;                           // 16 channel tiles of 32
  const int  m0  = mt * 64;
  const int  n0  = nt * 32;
  const half_t* pbase = p16  + (long)b * SS * SS;
  const half_t* vbase = vT16 + (long)b * CC * SS;

  v8f acc[8] = {};
  gemm4x2(pbase, SS, m0, vbase, SS, n0, SS, lane, acc);

  const int hi = lane >> 4;
  const int nl = lane & 15;
#pragma unroll
  for (int i = 0; i < 4; ++i) {
#pragma unroll
    for (int j = 0; j < 2; ++j) {
      const int n = n0 + j * 16 + nl;
      const v8f a = acc[i * 2 + j];
#pragma unroll
      for (int r = 0; r < 8; ++r) {
        long m = (long)b * SS + m0 + i * 16 + hi * 8 + r;
        o16[m * CC + n] = (half_t)a[r];
      }
    }
  }
}

// ---------------------------------------------------------------------------
// 7) out = O @ Wo^T + bo + residual, written f32 in [B, C, H, W] layout
//    (contiguous v4f stores: each lane owns 8 consecutive spatial positions)
// ---------------------------------------------------------------------------
__global__ __launch_bounds__(256) void proj_kernel(const half_t* __restrict__ o16,
                                                   const half_t* __restrict__ wo16,
                                                   const float* __restrict__ bo,
                                                   const float* __restrict__ x,
                                                   float* __restrict__ out) {
  const int lane = threadIdx.x & 31;
  const int wave = threadIdx.x >> 5;
  const long tile = (long)blockIdx.x * 8 + wave;       // < 8192
  const int  nt = (int)(tile & 15);
  const int  mt = (int)(tile >> 4);                    // 512 m-tiles of 64
  const int  m0 = mt * 64;
  const int  n0 = nt * 32;

  v8f acc[8] = {};
  gemm4x2(o16, CC, m0, wo16, CC, n0, CC, lane, acc);

  const int hi = lane >> 4;
  const int nl = lane & 15;
#pragma unroll
  for (int i = 0; i < 4; ++i) {
    const long m0i = (long)m0 + i * 16 + hi * 8;       // 8 consecutive tokens
    const int  b   = (int)(m0i >> 10);
    const int  s0  = (int)(m0i & 1023);
#pragma unroll
    for (int j = 0; j < 2; ++j) {
      const int   n    = n0 + j * 16 + nl;
      const float bias = bo[n];
      const long  off  = ((long)b * CC + n) * SS + s0;
      const v8f   a    = acc[i * 2 + j];
      v4f x0 = *(const v4f*)(x + off);
      v4f x1 = *(const v4f*)(x + off + 4);
      v4f o0, o1;
#pragma unroll
      for (int r = 0; r < 4; ++r) {
        o0[r] = a[r]     + bias + x0[r];
        o1[r] = a[4 + r] + bias + x1[r];
      }
      *(v4f*)(out + off)     = o0;
      *(v4f*)(out + off + 4) = o1;
    }
  }
}

// ---------------------------------------------------------------------------
extern "C" void kernel_launch(void* const* d_in, const int* in_sizes, int n_in,
                              void* d_out, int out_size, void* d_ws, size_t ws_size,
                              hipStream_t stream) {
  const float* x  = (const float*)d_in[0];
  const float* gw = (const float*)d_in[1];
  const float* gb = (const float*)d_in[2];
  const float* Wq = (const float*)d_in[3];
  const float* bq = (const float*)d_in[4];
  const float* Wk = (const float*)d_in[5];
  const float* bk = (const float*)d_in[6];
  const float* Wv = (const float*)d_in[7];
  const float* bv = (const float*)d_in[8];
  const float* Wo = (const float*)d_in[9];
  const float* bo = (const float*)d_in[10];
  float* out = (float*)d_out;

  char* ws = (char*)d_ws;
  // Workspace layout (bytes):
  //   [0,   2 MB)  Wq/Wk/Wv/Wo in f16
  //   [2,  34 MB)  t16  (GroupNorm output, later reused as attention O)
  //   [34, 66 MB)  q16
  //   [66, 98 MB)  k16
  //   [98,130 MB)  vT16 (V transposed [B,C,S])
  //   [130,194 MB) scores/probs f16 [B,S,S]
  half_t* w16  = (half_t*)(ws);
  half_t* t16  = (half_t*)(ws + (2l   << 20));
  half_t* q16  = (half_t*)(ws + (34l  << 20));
  half_t* k16  = (half_t*)(ws + (66l  << 20));
  half_t* vT16 = (half_t*)(ws + (98l  << 20));
  half_t* p16  = (half_t*)(ws + (130l << 20));
  half_t* o16  = t16;  // reuse: t is dead after the QKV GEMM

  wcvt_kernel   <<<dim3(1024, 4), 256, 0, stream>>>(Wq, Wk, Wv, Wo, w16);
  gnorm_kernel  <<<BB * NG,       256, 0, stream>>>(x, gw, gb, t16);
  qkv_kernel    <<<3072,          256, 0, stream>>>(t16, w16, bq, bk, bv, q16, k16, vT16);
  scores_kernel <<<2048,          256, 0, stream>>>(q16, k16, p16);
  softmax_kernel<<<BB * SS,       256, 0, stream>>>(p16);
  pv_kernel     <<<1024,          256, 0, stream>>>(p16, vT16, o16);
  proj_kernel   <<<1024,          256, 0, stream>>>(o16, w16 + 3l * CC * CC, bo, x, out);
}
